// LSTMbackbone_6579889897820
// MI455X (gfx1250) — matrix operand order repaired and checked by reading, not compiled, not run on hardware
//
#include <hip/hip_runtime.h>
#include <hip/hip_bf16.h>

// ---------------------------------------------------------------------------
// Fused 2-layer LSTM, persistent per-batch-tile kernel for gfx1250 (MI455X).
//   grid  = B/16 workgroups, each owns 16 batch rows for the whole T loop
//   block = 320 threads = 10 wave32; each wave owns 2 N-tiles (32 gate cols)
//   GEMMs via v_wmma_f32_16x16x32_f16, weights persistent in VGPR fragments
//   X stream software-pipelined: double-buffered LDS + 1-iteration-deep
//   register prefetch + global_prefetch 16 steps ahead (no load stalls).
// ---------------------------------------------------------------------------

typedef _Float16 v16h __attribute__((ext_vector_type(16)));
typedef _Float16 v8h  __attribute__((ext_vector_type(8)));
typedef float    v8f  __attribute__((ext_vector_type(8)));

#define B_      2048
#define T_      512
#define D_      8
#define H_      80
#define G_      320      // 4*H
#define BT      16       // batch tile (WMMA M)
#define THREADS 320      // 10 wave32
#define XLD     40       // xbuf leading dim (halves), 16B aligned, bank-skewed
#define HLD     104      // h buf leading dim (halves)
#define GLD     328      // gates buf leading dim (floats), bank-skewed

static __device__ __forceinline__ float sigf(float x) {
  return 1.0f / (1.0f + __expf(-x));
}
static __device__ __forceinline__ float tanh_fast(float x) {
  float e = __expf(-2.0f * fabsf(x));
  float t = (1.0f - e) / (1.0f + e);
  return (x < 0.0f) ? -t : t;
}

static __device__ __forceinline__ v8f wmma_f16(v16h a, v16h b, v8f c) {
  // D = A(16x32,f16) * B(32x16,f16) + C(16x16,f32)
  return __builtin_amdgcn_wmma_f32_16x16x32_f16(false, a, false, b,
                                                (short)0, c, false, false);
}

// B-matrix fragment (32x16, K-major per lane): lane<16 holds K=0..7 & 16..23 of
// column n=lane; lane>=16 holds K=8..15 & 24..31 of column n=lane-16.
// Source W is row-major [4H x Kdim] f32; B(k,n) = W[n,k]; K padded w/ zeros.
static __device__ __forceinline__ v16h load_bfrag(const float* __restrict__ W,
                                                  int Kdim, int nT, int kt,
                                                  int lane) {
  const int n  = nT * 16 + (lane & 15);
  const int kb = kt * 32 + ((lane & 16) ? 8 : 0);
  const float* row = W + n * Kdim;
  v16h r;
#pragma unroll
  for (int i = 0; i < 16; ++i) {
    int k = kb + (i < 8 ? i : i + 8);
    r[i] = (_Float16)((k < Kdim) ? row[k] : 0.0f);
  }
  return r;
}

// A-matrix fragment (16x32): lane m=lane%16; lanes<16 take K 0..7 & 16..23,
// lanes>=16 take K 8..15 & 24..31 (per ISA 16-bit A layout). Two 16B LDS loads.
static __device__ __forceinline__ v16h load_afrag(const _Float16* buf, int ld,
                                                  int kt, int lane) {
  const int m   = lane & 15;
  const int off = (lane & 16) ? 8 : 0;
  const _Float16* p = buf + m * ld + kt * 32 + off;
  v8h lo = *(const v8h*)(p);
  v8h hi = *(const v8h*)(p + 16);
  v16h a;
#pragma unroll
  for (int i = 0; i < 8; ++i) { a[i] = lo[i]; a[i + 8] = hi[i]; }
  return a;
}

__global__ __launch_bounds__(THREADS)
void lstm2_fused_wmma(const float* __restrict__ X,
                      const float* __restrict__ Wih0, const float* __restrict__ Whh0,
                      const float* __restrict__ bih0, const float* __restrict__ bhh0,
                      const float* __restrict__ Wih1, const float* __restrict__ Whh1,
                      const float* __restrict__ bih1, const float* __restrict__ bhh1,
                      const int* __restrict__ slen,
                      float* __restrict__ out) {
  __shared__ _Float16 xbuf[2][BT * XLD]; // double-buffered x_t tile, f16;
                                         // cols 8..31 stay zero (K pad)
  __shared__ _Float16 h1buf[BT * HLD];   // h1, f16, cols 80..95 zero (K pad)
  __shared__ _Float16 h2buf[BT * HLD];   // h2, f16
  __shared__ float    gbuf[BT * GLD];    // f32 gate accumulators [16 x 320]
  __shared__ float    bias1[G_];
  __shared__ float    bias2[G_];
  __shared__ int      lenbuf[BT];

  const int tid   = threadIdx.x;
  const int lane  = tid & 31;
  const int wv    = tid >> 5;            // wave 0..9 -> N-tiles 2wv, 2wv+1
  const int bBase = blockIdx.x * BT;

  // ---- one-time init ------------------------------------------------------
  for (int i = tid; i < 2 * BT * XLD; i += THREADS) xbuf[0][i] = (_Float16)0.0f;
  for (int i = tid; i < BT * HLD; i += THREADS) {
    h1buf[i] = (_Float16)0.0f;
    h2buf[i] = (_Float16)0.0f;
  }
  bias1[tid] = bih0[tid] + bhh0[tid];
  bias2[tid] = bih1[tid] + bhh1[tid];
  if (tid < BT) lenbuf[tid] = slen[bBase + tid];

  // persistent weight fragments in VGPRs (loaded once from HBM)
  v16h bIh0[2], bHh0[2][3], bIh1[2][3], bHh1[2][3];
#pragma unroll
  for (int j = 0; j < 2; ++j) {
    const int nT = 2 * wv + j;
    bIh0[j] = load_bfrag(Wih0, D_, nT, 0, lane);
#pragma unroll
    for (int kt = 0; kt < 3; ++kt) {
      bHh0[j][kt] = load_bfrag(Whh0, H_, nT, kt, lane);
      bIh1[j][kt] = load_bfrag(Wih1, H_, nT, kt, lane);
      bHh1[j][kt] = load_bfrag(Whh1, H_, nT, kt, lane);
    }
  }

  // X staging pipeline: threads 0..127 own one (row,d) element of the tile.
  const int xr = tid >> 3;               // row 0..15
  const int xd = tid & 7;                // d   0..7
  const float* xptr = X + (size_t)(bBase + xr) * T_ * D_ + xd;
  float xv = 0.0f;                       // holds x_{t+1} during iteration t
  if (tid < BT * D_) {
    xbuf[0][xr * XLD + xd] = (_Float16)xptr[0];   // x_0
    xv = xptr[D_];                                // x_1 (T_ >= 2)
  }

  // elementwise ownership: 4 contiguous cols of one row per thread (4 | 80)
  const int erow = tid / 20;
  const int ecol = (tid % 20) * 4;
  float c1[4] = {0.f, 0.f, 0.f, 0.f};
  float c2[4] = {0.f, 0.f, 0.f, 0.f};

  __syncthreads();
  const int mylen = lenbuf[erow];

  // ---- time recurrence (4 barriers / step) --------------------------------
  for (int t = 0; t < T_; ++t) {
    // ---- layer 1: gates = x_t @ Wih0^T + h1 @ Whh0^T ----------------------
    {
      v16h aX = load_afrag(xbuf[t & 1], XLD, 0, lane);
      v16h aH[3];
#pragma unroll
      for (int kt = 0; kt < 3; ++kt) aH[kt] = load_afrag(h1buf, HLD, kt, lane);
#pragma unroll
      for (int j = 0; j < 2; ++j) {
        v8f acc = {0.f, 0.f, 0.f, 0.f, 0.f, 0.f, 0.f, 0.f};
        acc = wmma_f16(aX, bIh0[j], acc);
#pragma unroll
        for (int kt = 0; kt < 3; ++kt) acc = wmma_f16(aH[kt], bHh0[j][kt], acc);
        const int n  = (2 * wv + j) * 16 + (lane & 15);
        const int mb = (lane & 16) ? 8 : 0;
#pragma unroll
        for (int r = 0; r < 8; ++r) gbuf[(mb + r) * GLD + n] = acc[r];
      }
    }
    __syncthreads();

    // ---- layer 1 elementwise (c1 in registers, h1 -> LDS f16) -------------
    {
      const float* gr = gbuf + erow * GLD;
#pragma unroll
      for (int u = 0; u < 4; ++u) {
        const int col = ecol + u;
        float ig = gr[col]          + bias1[col];
        float fg = gr[H_ + col]     + bias1[H_ + col];
        float gg = gr[2 * H_ + col] + bias1[2 * H_ + col];
        float og = gr[3 * H_ + col] + bias1[3 * H_ + col];
        c1[u] = sigf(fg) * c1[u] + sigf(ig) * tanh_fast(gg);
        float h = sigf(og) * tanh_fast(c1[u]);
        h1buf[erow * HLD + col] = (_Float16)h;
      }
    }
    __syncthreads();

    // ---- layer 2: gates = h1 @ Wih1^T + h2 @ Whh1^T -----------------------
    {
      v16h aH1[3], aH2[3];
#pragma unroll
      for (int kt = 0; kt < 3; ++kt) {
        aH1[kt] = load_afrag(h1buf, HLD, kt, lane);
        aH2[kt] = load_afrag(h2buf, HLD, kt, lane);
      }
#pragma unroll
      for (int j = 0; j < 2; ++j) {
        v8f acc = {0.f, 0.f, 0.f, 0.f, 0.f, 0.f, 0.f, 0.f};
#pragma unroll
        for (int kt = 0; kt < 3; ++kt) acc = wmma_f16(aH1[kt], bIh1[j][kt], acc);
#pragma unroll
        for (int kt = 0; kt < 3; ++kt) acc = wmma_f16(aH2[kt], bHh1[j][kt], acc);
        const int n  = (2 * wv + j) * 16 + (lane & 15);
        const int mb = (lane & 16) ? 8 : 0;
#pragma unroll
        for (int r = 0; r < 8; ++r) gbuf[(mb + r) * GLD + n] = acc[r];
      }
    }
    __syncthreads();

    // ---- layer 2 elementwise + packed-seq output --------------------------
    {
      const float* gr = gbuf + erow * GLD;
#pragma unroll
      for (int u = 0; u < 4; ++u) {
        const int col = ecol + u;
        float ig = gr[col]          + bias2[col];
        float fg = gr[H_ + col]     + bias2[H_ + col];
        float gg = gr[2 * H_ + col] + bias2[2 * H_ + col];
        float og = gr[3 * H_ + col] + bias2[3 * H_ + col];
        c2[u] = sigf(fg) * c2[u] + sigf(ig) * tanh_fast(gg);
        float h = sigf(og) * tanh_fast(c2[u]);
        h2buf[erow * HLD + col] = (_Float16)h;
        if (t == mylen - 1)
          out[(size_t)(bBase + erow) * H_ + col] = h;
      }
    }

    // ---- pipelined X staging for t+1 (overlaps with elementwise) ----------
    // xbuf[(t+1)&1] is not read until GEMM1 of iteration t+1 (after the
    // barrier below); xv was loaded a full iteration ago, so no load stall.
    if (tid < BT * D_) {
      xbuf[(t + 1) & 1][xr * XLD + xd] = (_Float16)xv;      // x_{t+1}
      if (t + 2 < T_) xv = xptr[(size_t)(t + 2) * D_];      // issue x_{t+2}
      else            xv = 0.0f;
      if (t + 16 < T_)                                      // warm L2/L0
        __builtin_prefetch(xptr + (size_t)(t + 16) * D_, 0, 0);
    }
    __syncthreads();
  }
}

extern "C" void kernel_launch(void* const* d_in, const int* in_sizes, int n_in,
                              void* d_out, int out_size, void* d_ws, size_t ws_size,
                              hipStream_t stream) {
  (void)in_sizes; (void)n_in; (void)out_size; (void)d_ws; (void)ws_size;
  const float* X    = (const float*)d_in[0];
  const float* Wih0 = (const float*)d_in[1];
  const float* Whh0 = (const float*)d_in[2];
  const float* bih0 = (const float*)d_in[3];
  const float* bhh0 = (const float*)d_in[4];
  const float* Wih1 = (const float*)d_in[5];
  const float* Whh1 = (const float*)d_in[6];
  const float* bih1 = (const float*)d_in[7];
  const float* bhh1 = (const float*)d_in[8];
  const int*   len  = (const int*)d_in[9];
  float*       out  = (float*)d_out;

  lstm2_fused_wmma<<<dim3(B_ / BT), dim3(THREADS), 0, stream>>>(
      X, Wih0, Whh0, bih0, bhh0, Wih1, Whh1, bih1, bhh1, len, out);
}
